// AttentionTest_14044543058050
// MI455X (gfx1250) — matrix operand order repaired and checked by reading, not compiled, not run on hardware
//
#include <hip/hip_runtime.h>
#include <cmath>

typedef __attribute__((ext_vector_type(16))) __bf16 v16bf;
typedef __attribute__((ext_vector_type(8)))  __bf16 v8bf;
typedef __attribute__((ext_vector_type(8)))  float  v8f;

#define EPSILON_F 0.01f

__device__ __forceinline__ float selu_f(float x) {
    const float alpha = 1.6732632423543772f;
    const float scale = 1.0507009873554805f;
    return x > 0.0f ? scale * x : scale * alpha * (expf(x) - 1.0f);
}

__device__ __forceinline__ float gelu_exact(float x) {
    return 0.5f * x * (1.0f + erff(x * 0.70710678118654752f));
}

// ---- CDNA5 async global->LDS copy (16B per lane), tracked with ASYNCcnt ----
// GLOBAL_LOAD_ASYNC_TO_LDS_B128: VDST = LDS byte offset VGPR, VADDR = 64-bit
// global address, GV mode (no SADDR). Generic shared pointers carry the
// workgroup-relative LDS offset in their low 32 bits (ISA §10.2 aperture map).
__device__ __forceinline__ void async_cp16(void* lds_dst, const void* gsrc) {
    unsigned lds_off = (unsigned)(reinterpret_cast<uintptr_t>(lds_dst));
    asm volatile("global_load_async_to_lds_b128 %0, %1, off"
                 :: "v"(lds_off), "v"(gsrc)
                 : "memory");
}

__device__ __forceinline__ void async_wait0() {
    asm volatile("s_wait_asynccnt 0" ::: "memory");
}

// ---------------- prep kernels ----------------

__global__ void cast_x_bf16(const float* __restrict__ x, __bf16* __restrict__ xb, int n) {
    int i = blockIdx.x * blockDim.x + threadIdx.x;
    if (i < n) xb[i] = (__bf16)x[i];
}

// Wqkv (4,512,1536) -> col-major bf16 for heads {0,3}: W1b[hs][n*512 + k]
__global__ void prep_w1(const float* __restrict__ Wqkv, __bf16* __restrict__ W1b) {
    int i = blockIdx.x * blockDim.x + threadIdx.x;
    if (i >= 2 * 1536 * 512) return;
    int hs  = i / (1536 * 512);
    int rem = i % (1536 * 512);
    int n = rem / 512, k = rem % 512;
    int h = hs ? 3 : 0;
    W1b[i] = (__bf16)Wqkv[(size_t)h * 512 * 1536 + (size_t)k * 1536 + n];
}

// Wp (4,512,512) -> col-major bf16 for heads {0,3}: Wpb[hs][p*512 + k]
__global__ void prep_wp(const float* __restrict__ Wp, __bf16* __restrict__ Wpb) {
    int i = blockIdx.x * blockDim.x + threadIdx.x;
    if (i >= 2 * 512 * 512) return;
    int hs  = i / (512 * 512);
    int rem = i % (512 * 512);
    int n = rem / 512, k = rem % 512;
    int h = hs ? 3 : 0;
    Wpb[i] = (__bf16)Wp[(size_t)h * 512 * 512 + (size_t)k * 512 + n];
}

// ---------------- fragment helpers (CDNA5 16-bit WMMA layouts) ----------------
// A 16x32: lane L -> row M=L%16; elements e<8: K = hf*8+e ; e>=8: K = 16+hf*8+(e-8)
// B 32x16: lane L -> col N=L%16; elements e<8: K = hf*8+e ; e>=8: K = 16+hf*8+(e-8)   (hf = L>>4)

__device__ __forceinline__ v16bf load_a_frag(const __bf16* row_ptr, int hf) {
    v16bf a;
#pragma unroll
    for (int e = 0; e < 8; ++e) {
        a[e]     = row_ptr[hf * 8 + e];
        a[e + 8] = row_ptr[16 + hf * 8 + e];
    }
    return a;
}

__device__ __forceinline__ v16bf load_b_frag(const __bf16* col_ptr, int hf) {
    v16bf b;
#pragma unroll
    for (int e = 0; e < 8; ++e) {
        b[e]     = col_ptr[hf * 8 + e];
        b[e + 8] = col_ptr[16 + hf * 8 + e];
    }
    return b;
}

// ---------------- QKV projection: selu(X @ W1 + b) ----------------
// M=32768 (b,s,n), K=512, N=1536. Outputs: Q row-major bf16; K,V transposed
// per 1024-token block: Ktb/Vtb[blk][d*1024 + n_tok]

__global__ __launch_bounds__(256)
void gemm_qkv(const __bf16* __restrict__ Xb, const __bf16* __restrict__ W1bh,
              const float* __restrict__ bias,
              __bf16* __restrict__ Qb, __bf16* __restrict__ Ktb, __bf16* __restrict__ Vtb) {
    __shared__ __bf16 lA[64 * 32];
    __shared__ __bf16 lB[64 * 32];
    const int row0 = blockIdx.y * 64;
    const int col0 = blockIdx.x * 64;
    const int t = threadIdx.x, lane = t & 31, w = t >> 5;
    const int tr = w & 3, tc0 = (w >> 2) * 2;
    const int hf = lane >> 4, l16 = lane & 15;

    v8f acc[2] = {{}, {}};
    for (int k0 = 0; k0 < 512; k0 += 32) {
        const int r = t >> 2, kk = (t & 3) * 8;
        async_cp16(&lA[r * 32 + kk], &Xb[(size_t)(row0 + r) * 512 + k0 + kk]);
        async_cp16(&lB[r * 32 + kk], &W1bh[(size_t)(col0 + r) * 512 + k0 + kk]);
        async_wait0();
        __syncthreads();
        v16bf a = load_a_frag(&lA[(tr * 16 + l16) * 32], hf);
#pragma unroll
        for (int j = 0; j < 2; ++j) {
            v16bf b = load_b_frag(&lB[((tc0 + j) * 16 + l16) * 32], hf);
            acc[j] = __builtin_amdgcn_wmma_f32_16x16x32_bf16(false, a, false, b,
                                                             (short)0, acc[j], false, false);
        }
        __syncthreads();
    }
#pragma unroll
    for (int j = 0; j < 2; ++j) {
        int n = col0 + (tc0 + j) * 16 + l16;
        float bv = bias[n];
#pragma unroll
        for (int r = 0; r < 8; ++r) {
            int m = row0 + tr * 16 + hf * 8 + r;
            __bf16 vb = (__bf16)selu_f(acc[j][r] + bv);
            if (n < 512) {
                Qb[(size_t)m * 512 + n] = vb;
            } else if (n < 1024) {
                int d = n - 512, blk = m >> 10, nn = m & 1023;
                Ktb[(size_t)blk * 524288 + (size_t)d * 1024 + nn] = vb;
            } else {
                int d = n - 1024, blk = m >> 10, nn = m & 1023;
                Vtb[(size_t)blk * 524288 + (size_t)d * 1024 + nn] = vb;
            }
        }
    }
}

// ---------------- kT @ v (per (b,s) block), scaled by 1/512 ----------------
// A = Ktb[blk] (512x1024 row-major), B = Vtb[blk] (col-major). Out transposed:
// KTVt[blk][col*512 + row] bf16 so it serves as col-major B for stage 3.

__global__ __launch_bounds__(256)
void gemm_ktv(const __bf16* __restrict__ Ktb, const __bf16* __restrict__ Vtb,
              __bf16* __restrict__ KTVt) {
    __shared__ __bf16 lA[64 * 32];
    __shared__ __bf16 lB[64 * 32];
    const int z = blockIdx.z;
    const size_t base = (size_t)z * 524288;
    const int row0 = blockIdx.y * 64;
    const int col0 = blockIdx.x * 64;
    const int t = threadIdx.x, lane = t & 31, w = t >> 5;
    const int tr = w & 3, tc0 = (w >> 2) * 2;
    const int hf = lane >> 4, l16 = lane & 15;

    v8f acc[2] = {{}, {}};
    for (int k0 = 0; k0 < 1024; k0 += 32) {
        const int r = t >> 2, kk = (t & 3) * 8;
        async_cp16(&lA[r * 32 + kk], &Ktb[base + (size_t)(row0 + r) * 1024 + k0 + kk]);
        async_cp16(&lB[r * 32 + kk], &Vtb[base + (size_t)(col0 + r) * 1024 + k0 + kk]);
        async_wait0();
        __syncthreads();
        v16bf a = load_a_frag(&lA[(tr * 16 + l16) * 32], hf);
#pragma unroll
        for (int j = 0; j < 2; ++j) {
            v16bf b = load_b_frag(&lB[((tc0 + j) * 16 + l16) * 32], hf);
            acc[j] = __builtin_amdgcn_wmma_f32_16x16x32_bf16(false, a, false, b,
                                                             (short)0, acc[j], false, false);
        }
        __syncthreads();
    }
#pragma unroll
    for (int j = 0; j < 2; ++j) {
        int n = col0 + (tc0 + j) * 16 + l16;
#pragma unroll
        for (int r = 0; r < 8; ++r) {
            int m = row0 + tr * 16 + hf * 8 + r;
            KTVt[(size_t)z * 262144 + (size_t)n * 512 + m] = (__bf16)(acc[j][r] * (1.0f / 512.0f));
        }
    }
}

// ---------------- fused: z = Q @ KTV ; softmax(row) ; gelu(att @ Wp + bp) ----------------
// One workgroup = 16-row strip x 512 cols. 8 waves x 4 WMMA tiles.

__global__ __launch_bounds__(256)
void attn_proj(const __bf16* __restrict__ Qb, const __bf16* __restrict__ KTVt,
               const __bf16* __restrict__ Wpbh, const float* __restrict__ bph,
               float* __restrict__ proj0, float* __restrict__ out, int head3) {
    __shared__ __bf16 lA[16 * 32];
    __shared__ __bf16 lB[512 * 32];
    __shared__ float  zb[16 * 512];
    __shared__ __bf16 attb[16 * 512];
    const int mBase = blockIdx.x * 16;
    const int blk = mBase >> 10;
    const int t = threadIdx.x, lane = t & 31, w = t >> 5;
    const int hf = lane >> 4, l16 = lane & 15;

    // ---- stage 1: z = Q_strip @ KTV_blk ----
    v8f acc[4] = {{}, {}, {}, {}};
    for (int k0 = 0; k0 < 512; k0 += 32) {
        if (t < 64) {
            int r = t >> 2, kk = (t & 3) * 8;
            async_cp16(&lA[r * 32 + kk], &Qb[(size_t)(mBase + r) * 512 + k0 + kk]);
        }
#pragma unroll
        for (int j = 0; j < 8; ++j) {
            int idx = j * 256 + t;                 // 0..2047 16B chunks
            int c = idx >> 2, kk = (idx & 3) * 8;
            async_cp16(&lB[c * 32 + kk],
                       &KTVt[(size_t)blk * 262144 + (size_t)c * 512 + k0 + kk]);
        }
        async_wait0();
        __syncthreads();
        v16bf a = load_a_frag(&lA[l16 * 32], hf);
#pragma unroll
        for (int j = 0; j < 4; ++j) {
            int col = (w * 4 + j) * 16 + l16;
            v16bf b = load_b_frag(&lB[col * 32], hf);
            acc[j] = __builtin_amdgcn_wmma_f32_16x16x32_bf16(false, a, false, b,
                                                             (short)0, acc[j], false, false);
        }
        __syncthreads();
    }
#pragma unroll
    for (int j = 0; j < 4; ++j) {
        int col = (w * 4 + j) * 16 + l16;
#pragma unroll
        for (int r = 0; r < 8; ++r) zb[(hf * 8 + r) * 512 + col] = acc[j][r];
    }
    __syncthreads();

    // ---- softmax over each row of 512 (wave32 shuffle reduction) ----
#pragma unroll
    for (int rr = 0; rr < 2; ++rr) {
        int row = w * 2 + rr;
        float mx = -3.4e38f;
#pragma unroll
        for (int i = 0; i < 16; ++i) mx = fmaxf(mx, zb[row * 512 + lane + i * 32]);
        for (int msk = 16; msk >= 1; msk >>= 1) mx = fmaxf(mx, __shfl_xor(mx, msk, 32));
        float ev[16], s = 0.0f;
#pragma unroll
        for (int i = 0; i < 16; ++i) {
            ev[i] = expf(zb[row * 512 + lane + i * 32] - mx);
            s += ev[i];
        }
        for (int msk = 16; msk >= 1; msk >>= 1) s += __shfl_xor(s, msk, 32);
        float inv = 1.0f / s;
#pragma unroll
        for (int i = 0; i < 16; ++i)
            attb[row * 512 + lane + i * 32] = (__bf16)(ev[i] * inv);
    }
    __syncthreads();

    // ---- stage 2: proj = att @ Wp ----
    v8f acc2[4] = {{}, {}, {}, {}};
    for (int k0 = 0; k0 < 512; k0 += 32) {
#pragma unroll
        for (int j = 0; j < 8; ++j) {
            int idx = j * 256 + t;
            int c = idx >> 2, kk = (idx & 3) * 8;
            async_cp16(&lB[c * 32 + kk], &Wpbh[(size_t)c * 512 + k0 + kk]);
        }
        async_wait0();
        __syncthreads();
        v16bf a;
#pragma unroll
        for (int e = 0; e < 8; ++e) {
            a[e]     = attb[l16 * 512 + k0 + hf * 8 + e];
            a[e + 8] = attb[l16 * 512 + k0 + 16 + hf * 8 + e];
        }
#pragma unroll
        for (int j = 0; j < 4; ++j) {
            int col = (w * 4 + j) * 16 + l16;
            v16bf b = load_b_frag(&lB[col * 32], hf);
            acc2[j] = __builtin_amdgcn_wmma_f32_16x16x32_bf16(false, a, false, b,
                                                              (short)0, acc2[j], false, false);
        }
        __syncthreads();
    }

    // ---- epilogue: gelu + bias ; head0 -> proj0 ; head3 -> + eps*proj0 + pose ----
#pragma unroll
    for (int j = 0; j < 4; ++j) {
        int p = (w * 4 + j) * 16 + l16;
        float bv = bph[p];
#pragma unroll
        for (int r = 0; r < 8; ++r) {
            int m = mBase + hf * 8 + r;
            float g = gelu_exact(acc2[j][r] + bv);
            size_t oi = (size_t)m * 512 + p;
            if (!head3) {
                proj0[oi] = g;
            } else {
                int pos = m & 1023;
                float den = 1000.0f * (float)(p & ~1) * (1.0f / 512.0f) + 0.01f;
                float ang = (float)pos / den;
                float pe = (p & 1) ? cosf(ang) : sinf(ang);
                out[oi] = g + EPSILON_F * proj0[oi] + pe;
            }
        }
    }
}

// ---------------- host ----------------

extern "C" void kernel_launch(void* const* d_in, const int* in_sizes, int n_in,
                              void* d_out, int out_size, void* d_ws, size_t ws_size,
                              hipStream_t stream) {
    (void)in_sizes; (void)n_in; (void)out_size; (void)ws_size;
    const float* x    = (const float*)d_in[0];
    const float* Wqkv = (const float*)d_in[1];
    const float* bqkv = (const float*)d_in[2];
    const float* Wp   = (const float*)d_in[3];
    const float* bp   = (const float*)d_in[4];
    float* out = (float*)d_out;

    char* ws = (char*)d_ws;
    size_t off = 0;
    auto take = [&](size_t bytes) { char* p = ws + off; off = (off + bytes + 255) & ~(size_t)255; return p; };

    __bf16* Xb   = (__bf16*)take((size_t)32768 * 512 * 2);        // x as bf16
    __bf16* W1b  = (__bf16*)take((size_t)2 * 1536 * 512 * 2);     // Wqkv[0,3] col-major
    __bf16* Wpb  = (__bf16*)take((size_t)2 * 512 * 512 * 2);      // Wp[0,3] col-major
    __bf16* Qb   = (__bf16*)take((size_t)32768 * 512 * 2);
    __bf16* Ktb  = (__bf16*)take((size_t)32768 * 512 * 2);        // per-block (512,1024)
    __bf16* Vtb  = (__bf16*)take((size_t)32768 * 512 * 2);
    __bf16* KTVt = (__bf16*)take((size_t)32 * 512 * 512 * 2);
    float*  proj0 = (float*)take((size_t)32768 * 512 * 4);

    cast_x_bf16<<<65536, 256, 0, stream>>>(x, Xb, 16777216);
    prep_w1<<<(2 * 1536 * 512 + 255) / 256, 256, 0, stream>>>(Wqkv, W1b);
    prep_wp<<<(2 * 512 * 512 + 255) / 256, 256, 0, stream>>>(Wp, Wpb);

    for (int hs = 0; hs < 2; ++hs) {            // heads 1,2 are dead code in the reference
        int h = hs ? 3 : 0;
        gemm_qkv<<<dim3(24, 512), 256, 0, stream>>>(
            Xb, W1b + (size_t)hs * 1536 * 512, bqkv + (size_t)h * 1536, Qb, Ktb, Vtb);
        gemm_ktv<<<dim3(8, 8, 32), 256, 0, stream>>>(Ktb, Vtb, KTVt);
        attn_proj<<<2048, 256, 0, stream>>>(
            Qb, KTVt, Wpb + (size_t)hs * 512 * 512, bp + (size_t)h * 512, proj0, out, hs);
    }
}
